// T5Attention_6940667150594
// MI455X (gfx1250) — matrix operand order repaired and checked by reading, not compiled, not run on hardware
//
#include <hip/hip_runtime.h>
#include <hip/hip_bf16.h>

typedef __attribute__((ext_vector_type(16))) __bf16 v16bf;
typedef __attribute__((ext_vector_type(8)))  __bf16 v8bf;
typedef __attribute__((ext_vector_type(8)))  float  v8f;
typedef __attribute__((ext_vector_type(4)))  int    v4i_;

#define SEQ     2048
#define DMODEL  1024
#define NHEAD   16
#define HDIM    64
#define MROWS   4096   // B*S

static __device__ __forceinline__ v16bf cat8(v8bf lo, v8bf hi) {
    return __builtin_shufflevector(lo, hi, 0,1,2,3,4,5,6,7,8,9,10,11,12,13,14,15);
}

// 16-lane DPP rotate (row_ror:N); row = 16 lanes matches the WMMA C-layout halves
template<int CTRL>
static __device__ __forceinline__ float dpp_rot(float x) {
    int i = __builtin_amdgcn_update_dpp(0, __float_as_int(x), CTRL, 0xf, 0xf, true);
    return __int_as_float(i);
}
static __device__ __forceinline__ float rowmax16(float v) {
    v = fmaxf(v, dpp_rot<0x121>(v));   // row_ror:1
    v = fmaxf(v, dpp_rot<0x122>(v));   // row_ror:2
    v = fmaxf(v, dpp_rot<0x124>(v));   // row_ror:4
    v = fmaxf(v, dpp_rot<0x128>(v));   // row_ror:8
    return v;
}
static __device__ __forceinline__ float rowsum16(float v) {
    v += dpp_rot<0x121>(v);
    v += dpp_rot<0x122>(v);
    v += dpp_rot<0x124>(v);
    v += dpp_rot<0x128>(v);
    return v;
}

// ---------------------------------------------------------------------------
// 1a) fp32 -> bf16 row-major copy (x)
// ---------------------------------------------------------------------------
__global__ void cvt_bf16_kernel(const float* __restrict__ src,
                                __bf16* __restrict__ dst, int n) {
    int i = (blockIdx.x * blockDim.x + threadIdx.x) * 4;
    if (i + 3 < n) {
        float4 v = *(const float4*)(src + i);
        dst[i + 0] = (__bf16)v.x;
        dst[i + 1] = (__bf16)v.y;
        dst[i + 2] = (__bf16)v.z;
        dst[i + 3] = (__bf16)v.w;
    }
}

// ---------------------------------------------------------------------------
// 1b) fp32 [K][N] -> bf16 transposed [N][K]; tile staged via async-to-LDS
//     grid (N/32, K/32), 256 threads
// ---------------------------------------------------------------------------
__global__ __launch_bounds__(256)
void cvt_bf16_t_kernel(const float* __restrict__ src, __bf16* __restrict__ dst) {
    __shared__ float tile[32][36];            // stride 144B: 16B-aligned rows
    const int n0 = blockIdx.x * 32;
    const int k0 = blockIdx.y * 32;

#if __has_builtin(__builtin_amdgcn_global_load_async_to_lds_b128)
    {
        const int q  = threadIdx.x & 7;       // 16B quad within row
        const int kr = threadIdx.x >> 3;      // 0..31 source rows
        const float* g = src + (size_t)(k0 + kr) * DMODEL + n0 + q * 4;
        __builtin_amdgcn_global_load_async_to_lds_b128(
            (__attribute__((address_space(1))) v4i_*)g,
            (__attribute__((address_space(3))) v4i_*)&tile[kr][q * 4], 0, 0);
    }
  #if __has_builtin(__builtin_amdgcn_s_wait_asynccnt)
    __builtin_amdgcn_s_wait_asynccnt(0);
  #else
    asm volatile("s_wait_asynccnt 0" ::: "memory");
  #endif
    __syncthreads();
#else
    {
        const int tx = threadIdx.x & 31, ty = threadIdx.x >> 5;
        #pragma unroll
        for (int i = 0; i < 32; i += 8)
            tile[ty + i][tx] = src[(size_t)(k0 + ty + i) * DMODEL + n0 + tx];
        __syncthreads();
    }
#endif
    const int tx = threadIdx.x & 31, ty = threadIdx.x >> 5;
    #pragma unroll
    for (int i = 0; i < 32; i += 8)
        dst[(size_t)(n0 + ty + i) * DMODEL + k0 + tx] = (__bf16)tile[tx][ty + i];
}

// ---------------------------------------------------------------------------
// 2) T5 relative-position bias -> d_out second region, layout [H][S][S]
// ---------------------------------------------------------------------------
__global__ void bias_kernel(const float* __restrict__ emb,   // [32 buckets][16 heads]
                            float* __restrict__ bias) {
    int idx = blockIdx.x * blockDim.x + threadIdx.x;   // q*S + k
    if (idx >= SEQ * SEQ) return;
    int q = idx >> 11;
    int k = idx & (SEQ - 1);
    int rel = k - q;
    int bucket = (rel > 0) ? 16 : 0;
    int rp = rel < 0 ? -rel : rel;
    int v;
    if (rp < 8) {
        v = rp;
    } else {
        float t = __logf((float)rp * 0.125f) * (8.0f / 2.7725887222397811f); // /log(16)
        int vi = 8 + (int)t;
        v = vi < 15 ? vi : 15;
    }
    bucket += v;
    const float* e = emb + bucket * NHEAD;
    #pragma unroll
    for (int h = 0; h < NHEAD; h++)
        bias[(size_t)h * SEQ * SEQ + idx] = e[h];
}

// ---------------------------------------------------------------------------
// 3/5) bf16 WMMA GEMM, B pre-transposed [N][K]; 32x64 tile per wave
//      MODE 0: bf16 row-major   MODE 1: bf16 transposed [N][M]   MODE 2: f32
// ---------------------------------------------------------------------------
template<int MODE>
__global__ __launch_bounds__(32)
void gemm_kernel(const __bf16* __restrict__ A, const __bf16* __restrict__ Wt,
                 __bf16* __restrict__ outb, float* __restrict__ outf) {
    const int lane = threadIdx.x;
    const int ncol = lane & 15;
    const int half = lane >> 4;
    const int n0 = blockIdx.x * 64;
    const int m0 = blockIdx.y * 32;

    v8f acc[2][4];
    #pragma unroll
    for (int mi = 0; mi < 2; mi++)
        #pragma unroll
        for (int ni = 0; ni < 4; ni++)
            acc[mi][ni] = (v8f){0.f,0.f,0.f,0.f,0.f,0.f,0.f,0.f};

    const __bf16* arow0 = A  + (size_t)(m0 + ncol) * DMODEL + half * 8;
    const __bf16* brow0 = Wt + (size_t)(n0 + ncol) * DMODEL + half * 8;

    for (int k0 = 0; k0 < DMODEL; k0 += 32) {
        v16bf a[2], bm[4];
        #pragma unroll
        for (int mi = 0; mi < 2; mi++) {
            const __bf16* p = arow0 + (size_t)mi * 16 * DMODEL + k0;
            a[mi] = cat8(*(const v8bf*)p, *(const v8bf*)(p + 16));
        }
        #pragma unroll
        for (int ni = 0; ni < 4; ni++) {
            const __bf16* p = brow0 + (size_t)ni * 16 * DMODEL + k0;
            bm[ni] = cat8(*(const v8bf*)p, *(const v8bf*)(p + 16));
        }
        #pragma unroll
        for (int mi = 0; mi < 2; mi++)
            #pragma unroll
            for (int ni = 0; ni < 4; ni++)
                acc[mi][ni] = __builtin_amdgcn_wmma_f32_16x16x32_bf16(
                    false, a[mi], false, bm[ni], (short)0, acc[mi][ni], false, false);
    }

    #pragma unroll
    for (int mi = 0; mi < 2; mi++) {
        #pragma unroll
        for (int ni = 0; ni < 4; ni++) {
            if (MODE == 1) {
                v8bf p;
                #pragma unroll
                for (int r = 0; r < 8; r++) p[r] = (__bf16)acc[mi][ni][r];
                __bf16* dst = outb + (size_t)(n0 + ni * 16 + ncol) * MROWS
                                   + m0 + mi * 16 + half * 8;
                *(v8bf*)dst = p;
            } else {
                #pragma unroll
                for (int r = 0; r < 8; r++) {
                    int m = m0 + mi * 16 + r + half * 8;
                    size_t o = (size_t)m * DMODEL + n0 + ni * 16 + ncol;
                    if (MODE == 2) outf[o] = acc[mi][ni][r];
                    else           outb[o] = (__bf16)acc[mi][ni][r];
                }
            }
        }
    }
}

// ---------------------------------------------------------------------------
// 4) flash attention: one wave per (q-tile 16, head, batch); 64-key chunks
// ---------------------------------------------------------------------------
__global__ __launch_bounds__(32)
void attn_kernel(const __bf16* __restrict__ Qb, const __bf16* __restrict__ Kb,
                 const __bf16* __restrict__ Vt, const float* __restrict__ bias,
                 __bf16* __restrict__ Ob) {
    __shared__ __bf16 ldsP[16 * 64];   // P tile, rows=q, cols=key-in-chunk

    const int lane = threadIdx.x;
    const int ncol = lane & 15;
    const int half = lane >> 4;
    const int qt = blockIdx.x;
    const int h  = blockIdx.y;
    const int b  = blockIdx.z;
    const int qrow0 = b * SEQ + qt * 16;
    const float* biash = bias + (size_t)h * SEQ * SEQ;

    // Q A-fragments (row = ncol), d-chunks [0..31] and [32..63]
    const __bf16* qptr = Qb + (size_t)(qrow0 + ncol) * DMODEL + h * HDIM + half * 8;
    v16bf qa0 = cat8(*(const v8bf*)(qptr +  0), *(const v8bf*)(qptr + 16));
    v16bf qa1 = cat8(*(const v8bf*)(qptr + 32), *(const v8bf*)(qptr + 48));

    v8f o[4];
    #pragma unroll
    for (int t = 0; t < 4; t++) o[t] = (v8f){0.f,0.f,0.f,0.f,0.f,0.f,0.f,0.f};
    float mrun[8], lrun[8];
    #pragma unroll
    for (int r = 0; r < 8; r++) { mrun[r] = -1e30f; lrun[r] = 0.f; }

    for (int key0 = 0; key0 < SEQ; key0 += 64) {
        v8f st[4];
        #pragma unroll
        for (int j = 0; j < 4; j++) {
            int kt = key0 + j * 16;
            const __bf16* kp = Kb + (size_t)(b * SEQ + kt + ncol) * DMODEL
                                  + h * HDIM + half * 8;
            v16bf kb0 = cat8(*(const v8bf*)(kp +  0), *(const v8bf*)(kp + 16));
            v16bf kb1 = cat8(*(const v8bf*)(kp + 32), *(const v8bf*)(kp + 48));
            v8f s = {0.f,0.f,0.f,0.f,0.f,0.f,0.f,0.f};
            s = __builtin_amdgcn_wmma_f32_16x16x32_bf16(
                false, qa0, false, kb0, (short)0, s, false, false);
            s = __builtin_amdgcn_wmma_f32_16x16x32_bf16(
                false, qa1, false, kb1, (short)0, s, false, false);
            #pragma unroll
            for (int r = 0; r < 8; r++) {
                int m = r + half * 8;
                s[r] += biash[(size_t)(qt * 16 + m) * SEQ + kt + ncol];
            }
            st[j] = s;
        }

        // online softmax over the 64-key chunk (DPP 16-lane butterflies)
        #pragma unroll
        for (int r = 0; r < 8; r++) {
            float v = fmaxf(fmaxf(st[0][r], st[1][r]), fmaxf(st[2][r], st[3][r]));
            v = rowmax16(v);
            float mnew = fmaxf(mrun[r], v);
            float p0 = __expf(st[0][r] - mnew);
            float p1 = __expf(st[1][r] - mnew);
            float p2 = __expf(st[2][r] - mnew);
            float p3 = __expf(st[3][r] - mnew);
            float ps = rowsum16((p0 + p1) + (p2 + p3));
            float corr = __expf(mrun[r] - mnew);
            lrun[r] = lrun[r] * corr + ps;
            mrun[r] = mnew;
            o[0][r] *= corr; o[1][r] *= corr; o[2][r] *= corr; o[3][r] *= corr;
            st[0][r] = p0; st[1][r] = p1; st[2][r] = p2; st[3][r] = p3;
        }

        // C-layout -> A-layout via LDS (16x64 bf16)
        __syncthreads();
        #pragma unroll
        for (int r = 0; r < 8; r++) {
            int m = r + half * 8;
            #pragma unroll
            for (int j = 0; j < 4; j++)
                ldsP[m * 64 + j * 16 + ncol] = (__bf16)st[j][r];
        }
        __syncthreads();
        const __bf16* pp = &ldsP[ncol * 64 + half * 8];
        v16bf pa0 = cat8(*(const v8bf*)(pp +  0), *(const v8bf*)(pp + 16)); // keys 0..31
        v16bf pa1 = cat8(*(const v8bf*)(pp + 32), *(const v8bf*)(pp + 48)); // keys 32..63

        // O += P * V : V^T fragments contiguous along keys -> b128 loads
        #pragma unroll
        for (int t = 0; t < 4; t++) {
            const __bf16* vp = Vt + (size_t)(h * HDIM + t * 16 + ncol) * MROWS
                                  + b * SEQ + key0 + half * 8;
            v16bf vb0 = cat8(*(const v8bf*)(vp +  0), *(const v8bf*)(vp + 16));
            v16bf vb1 = cat8(*(const v8bf*)(vp + 32), *(const v8bf*)(vp + 48));
            o[t] = __builtin_amdgcn_wmma_f32_16x16x32_bf16(
                false, pa0, false, vb0, (short)0, o[t], false, false);
            o[t] = __builtin_amdgcn_wmma_f32_16x16x32_bf16(
                false, pa1, false, vb1, (short)0, o[t], false, false);
        }
    }

    // normalize and store (bf16 for the final GEMM)
    #pragma unroll
    for (int t = 0; t < 4; t++) {
        #pragma unroll
        for (int r = 0; r < 8; r++) {
            int m = r + half * 8;
            Ob[(size_t)(qrow0 + m) * DMODEL + h * HDIM + t * 16 + ncol] =
                (__bf16)(o[t][r] / lrun[r]);
        }
    }
}

// ---------------------------------------------------------------------------
// host
// ---------------------------------------------------------------------------
extern "C" void kernel_launch(void* const* d_in, const int* in_sizes, int n_in,
                              void* d_out, int out_size, void* d_ws, size_t ws_size,
                              hipStream_t stream) {
    const float* x   = (const float*)d_in[0];
    const float* Wq  = (const float*)d_in[1];
    const float* Wk  = (const float*)d_in[2];
    const float* Wv  = (const float*)d_in[3];
    const float* Wo  = (const float*)d_in[4];
    const float* emb = (const float*)d_in[5];

    float* out  = (float*)d_out;                       // [4096,1024]
    float* bias = out + (size_t)MROWS * DMODEL;        // [16,2048,2048]

    char* ws = (char*)d_ws;
    __bf16* xb   = (__bf16*)(ws + 0);                  //  8 MB
    __bf16* wqt  = (__bf16*)(ws +  8u * 1024 * 1024);  //  2 MB each, transposed [N][K]
    __bf16* wkt  = (__bf16*)(ws + 10u * 1024 * 1024);
    __bf16* wvt  = (__bf16*)(ws + 12u * 1024 * 1024);
    __bf16* wot  = (__bf16*)(ws + 14u * 1024 * 1024);
    __bf16* qb   = (__bf16*)(ws + 16u * 1024 * 1024);  //  8 MB, [M][N]
    __bf16* kb   = (__bf16*)(ws + 24u * 1024 * 1024);  //  8 MB, [M][N]
    __bf16* vt   = (__bf16*)(ws + 32u * 1024 * 1024);  //  8 MB, transposed [N][M]
    __bf16* ab   = (__bf16*)(ws + 40u * 1024 * 1024);  //  8 MB, end = 48 MB

    // 1) conversions (+ weight transposes)
    cvt_bf16_kernel<<<(MROWS * DMODEL) / 1024, 256, 0, stream>>>(x, xb, MROWS * DMODEL);
    dim3 tgrid(DMODEL / 32, DMODEL / 32);
    cvt_bf16_t_kernel<<<tgrid, 256, 0, stream>>>(Wq, wqt);
    cvt_bf16_t_kernel<<<tgrid, 256, 0, stream>>>(Wk, wkt);
    cvt_bf16_t_kernel<<<tgrid, 256, 0, stream>>>(Wv, wvt);
    cvt_bf16_t_kernel<<<tgrid, 256, 0, stream>>>(Wo, wot);

    // 2) relative-position bias (second output region; also read by attention)
    bias_kernel<<<(SEQ * SEQ) / 256, 256, 0, stream>>>(emb, bias);

    // 3) Q/K/V projections (WMMA, 32x64 tile per wave)
    dim3 ggrid(DMODEL / 64, MROWS / 32);
    gemm_kernel<0><<<ggrid, 32, 0, stream>>>(xb, wqt, qb, nullptr);
    gemm_kernel<0><<<ggrid, 32, 0, stream>>>(xb, wkt, kb, nullptr);
    gemm_kernel<1><<<ggrid, 32, 0, stream>>>(xb, wvt, vt, nullptr);  // V stored [N][M]

    // 4) flash attention (64-key chunks)
    dim3 agrid(SEQ / 16, NHEAD, 2);
    attn_kernel<<<agrid, 32, 0, stream>>>(qb, kb, vt, bias, ab);

    // 5) output projection -> f32 d_out
    gemm_kernel<2><<<ggrid, 32, 0, stream>>>(ab, wot, nullptr, out);
}
